// RSNN_merged_hidden_output_88845693485077
// MI455X (gfx1250) — matrix operand order repaired and checked by reading, not compiled, not run on hardware
//
#include <hip/hip_runtime.h>

// ---------------------------------------------------------------------------
// RSNN merged hidden/output population, CDNA5 (gfx1250) wave32 WMMA version.
//
//  * WT[m][k] f16 (m<2048, k<2560): k<512 -> W_fc[m][k];
//    k>=512 -> (recurrent*mask*binary)^T so every WMMA B fragment is one
//    contiguous 32B load per lane. WT (10 MB) stays resident in the 192MB L2
//    across all 100 steps.
//  * Single persistent kernel: t-loop inside, device-wide generation barrier
//    (atomic counter + s_sleep spin + device-scope fences) between steps —
//    removes 100 kernel-launch/drain latencies from the sequential scan.
//  * Per step: C[256x2048] = A[256x2560] * WT^T via v_wmma_f32_16x16x32_f16,
//    A rows = x16 (k<512) then f16 spikes (exact 0/1). LIF epilogue fused.
//  * Spikes double-buffered (f16); mem is owner-written only (no race).
// ---------------------------------------------------------------------------

typedef __attribute__((ext_vector_type(16))) _Float16 v16h;
typedef __attribute__((ext_vector_type(8)))  _Float16 v8h;
typedef __attribute__((ext_vector_type(8)))  float    v8f;

#define N_TOT   2048
#define NIN     512
#define KTOT    2560         // 512 ff + 2048 recurrent
#define BATCH   256
#define TSTEPS  100
#define NOUT    48
#define TAUF    0.96f
#define NBLOCKS 128          // 16 neuron tiles (x128) * 8 batch tiles (x32)

// --------------------------- preprocessing kernels -------------------------

__global__ void rsnn_prep_weights(const float* __restrict__ W_fc,
                                  const float* __restrict__ recur,
                                  const float* __restrict__ mask,
                                  const float* __restrict__ binary,
                                  _Float16* __restrict__ WT) {
    int idx = blockIdx.x * blockDim.x + threadIdx.x;    // over 2048*2560
    if (idx >= N_TOT * KTOT) return;
    int m = idx / KTOT;
    int k = idx - m * KTOT;
    float v;
    if (k < NIN) {
        v = W_fc[m * NIN + k];
    } else {
        int kk = k - NIN;                               // source row of eff_rec
        int j  = kk * N_TOT + m;                        // transpose gather
        v = recur[j] * mask[j] * binary[j];
    }
    WT[idx] = (_Float16)v;
}

__global__ void rsnn_prep_x(const float* __restrict__ x,
                            _Float16* __restrict__ x16, int n) {
    int idx = blockIdx.x * blockDim.x + threadIdx.x;
    if (idx < n) x16[idx] = (_Float16)x[idx];
}

__global__ void rsnn_init_state(float* __restrict__ mem,
                                _Float16* __restrict__ spk, int n,
                                unsigned* __restrict__ barrier_cnt) {
    int idx = blockIdx.x * blockDim.x + threadIdx.x;
    if (idx == 0) *barrier_cnt = 0u;
    if (idx < n) { mem[idx] = 0.0f; spk[idx] = (_Float16)0.0f; }
}

// --------------------------- persistent step kernel ------------------------
// 128 blocks x 128 threads (4 waves). Block tile 32(M batch) x 128(N neuron);
// wave tile 16 x 64 (4 f32 accumulators).

__global__ __launch_bounds__(128)
void rsnn_persistent(const _Float16* __restrict__ x16,  // [B, T, 512]
                     const _Float16* __restrict__ WT,   // [2048, 2560]
                     _Float16* __restrict__ spkA,       // [B, 2048]
                     _Float16* __restrict__ spkB,       // [B, 2048]
                     float* __restrict__ mem,           // [B, 2048]
                     float* __restrict__ out,           // [B, T, 48]
                     unsigned* __restrict__ barrier_cnt) {
    const int lane = threadIdx.x & 31;
    const int wave = threadIdx.x >> 5;
    const int wm   = wave >> 1;          // 0..1  wave row in block
    const int wn   = wave & 1;           // 0..1  wave col in block
    const int g    = lane >> 4;          // half-wave group
    const int r    = lane & 15;

    const int bn = blockIdx.x & 15;      // neuron block 0..15
    const int bm = blockIdx.x >> 4;      // batch block 0..7
    const int m_base = bm * 32 + wm * 16;
    const int n_base = bn * 128 + wn * 64;

    for (int t = 0; t < TSTEPS; ++t) {
        const _Float16* __restrict__ spike_in  = (t & 1) ? spkB : spkA;
        _Float16* __restrict__       spike_out = (t & 1) ? spkA : spkB;

        v8f acc[4];
#pragma unroll
        for (int j = 0; j < 4; ++j)
#pragma unroll
            for (int i = 0; i < 8; ++i) acc[j][i] = 0.0f;

        // ---- phase 1: feed-forward K = 0..511, A rows from x16[b][t][:] ----
        {
            const _Float16* arow =
                x16 + (size_t)(m_base + r) * (TSTEPS * NIN) + (size_t)t * NIN;
            for (int kk = 0; kk < NIN; kk += 32) {
                // A frag: group 0 -> K {kk..kk+15}
                //         group 1 -> K {kk+16..kk+31}
                v16h a = *(const v16h*)(arow + kk + g * 16);
#pragma unroll
                for (int j = 0; j < 4; ++j) {
                    const _Float16* bp =
                        WT + (size_t)(n_base + j * 16 + r) * KTOT + kk + g * 16;
                    v16h b = *(const v16h*)bp;
                    acc[j] = __builtin_amdgcn_wmma_f32_16x16x32_f16(
                        false, a, false, b, (short)0, acc[j], false, false);
                }
            }
        }

        // ---- phase 2: recurrent K = 0..2047, A rows from spike_in[b][:] ----
        {
            const _Float16* srow = spike_in + (size_t)(m_base + r) * N_TOT;
            for (int kk = 0; kk < N_TOT; kk += 32) {
                v16h a = *(const v16h*)(srow + kk + g * 16);
#pragma unroll
                for (int j = 0; j < 4; ++j) {
                    const _Float16* bp =
                        WT + (size_t)(n_base + j * 16 + r) * KTOT +
                        (NIN + kk) + g * 16;
                    v16h b = *(const v16h*)bp;
                    acc[j] = __builtin_amdgcn_wmma_f32_16x16x32_f16(
                        false, a, false, b, (short)0, acc[j], false, false);
                }
            }
        }

        // ---- LIF epilogue: mem = tau*mem*(1-s) + cur ; s' = (mem >= 1) ----
        // C/D layout: VGPR i holds (M = i + 8*g, N = r) for this lane.
#pragma unroll
        for (int j = 0; j < 4; ++j) {
            const int n = n_base + j * 16 + r;
#pragma unroll
            for (int i = 0; i < 8; ++i) {
                const int brow = m_base + i + 8 * g;
                const size_t idx = (size_t)brow * N_TOT + n;
                const float cur = acc[j][i];
                const float mo  = mem[idx];
                const float so  = (float)spike_in[idx];
                const float mn  = TAUF * mo * (1.0f - so) + cur;
                const float sn  = (mn >= 1.0f) ? 1.0f : 0.0f;
                mem[idx] = mn;
                spike_out[idx] = (_Float16)sn;
                if (n >= N_TOT - NOUT) {
                    out[(size_t)brow * (TSTEPS * NOUT) + (size_t)t * NOUT +
                        (n - (N_TOT - NOUT))] = sn;
                }
            }
        }

        // ---- device-wide generation barrier between timesteps ----
        __threadfence();                      // release our spike/mem stores
        __syncthreads();
        if (threadIdx.x == 0) {
            __hip_atomic_fetch_add(barrier_cnt, 1u, __ATOMIC_ACQ_REL,
                                   __HIP_MEMORY_SCOPE_AGENT);
            const unsigned target = (unsigned)NBLOCKS * (unsigned)(t + 1);
            while (__hip_atomic_load(barrier_cnt, __ATOMIC_ACQUIRE,
                                     __HIP_MEMORY_SCOPE_AGENT) < target) {
                __builtin_amdgcn_s_sleep(2);
            }
        }
        __syncthreads();
        __threadfence();                      // acquire: invalidate stale lines
    }
}

// ------------------------------- launcher ----------------------------------

extern "C" void kernel_launch(void* const* d_in, const int* in_sizes, int n_in,
                              void* d_out, int out_size, void* d_ws, size_t ws_size,
                              hipStream_t stream) {
    const float* x      = (const float*)d_in[0];   // [256,100,512]
    const float* W_fc   = (const float*)d_in[1];   // [2048,512]
    const float* recur  = (const float*)d_in[2];   // [2048,2048]
    const float* maskp  = (const float*)d_in[3];   // [2048,2048]
    const float* binary = (const float*)d_in[4];   // [2048,2048]
    float* out = (float*)d_out;                    // [256,100,48]

    // workspace layout (all offsets 256B aligned)
    char* ws = (char*)d_ws;
    const size_t WT_BYTES  = (size_t)N_TOT * KTOT * 2;         // 10,485,760
    const size_t X16_BYTES = (size_t)BATCH * TSTEPS * NIN * 2; // 26,214,400
    const size_t MEM_BYTES = (size_t)BATCH * N_TOT * 4;        //  2,097,152
    const size_t SPK_BYTES = (size_t)BATCH * N_TOT * 2;        //  1,048,576
    _Float16* WT   = (_Float16*)ws;
    _Float16* x16  = (_Float16*)(ws + WT_BYTES);
    float*    mem  = (float*)   (ws + WT_BYTES + X16_BYTES);
    _Float16* spkA = (_Float16*)(ws + WT_BYTES + X16_BYTES + MEM_BYTES);
    _Float16* spkB = (_Float16*)(ws + WT_BYTES + X16_BYTES + MEM_BYTES + SPK_BYTES);
    unsigned* bcnt = (unsigned*)(ws + WT_BYTES + X16_BYTES + MEM_BYTES + 2 * SPK_BYTES);

    {
        int n = N_TOT * KTOT;
        rsnn_prep_weights<<<(n + 255) / 256, 256, 0, stream>>>(
            W_fc, recur, maskp, binary, WT);
    }
    {
        int n = BATCH * TSTEPS * NIN;
        rsnn_prep_x<<<(n + 255) / 256, 256, 0, stream>>>(x, x16, n);
    }
    {
        int n = BATCH * N_TOT;
        rsnn_init_state<<<(n + 255) / 256, 256, 0, stream>>>(mem, spkA, n, bcnt);
    }

    rsnn_persistent<<<NBLOCKS, 128, 0, stream>>>(
        x16, WT, spkA, spkB, mem, out, bcnt);
}
